// TEN_MultiModal_33114197852605
// MI455X (gfx1250) — matrix operand order repaired and checked by reading, not compiled, not run on hardware
//
#include <hip/hip_runtime.h>
#include <math.h>

typedef __attribute__((ext_vector_type(16))) _Float16 v16h;
typedef __attribute__((ext_vector_type(8)))  _Float16 h8;
typedef __attribute__((ext_vector_type(8)))  float    v8f;

// ---------------------------------------------------------------------------
// Block-wide 2-value sum reduction (blockDim.x == 256, wave32)
// ---------------------------------------------------------------------------
__device__ __forceinline__ void blockReduce2(float& a, float& b, float* sbuf) {
  for (int o = 16; o > 0; o >>= 1) { a += __shfl_xor(a, o); b += __shfl_xor(b, o); }
  int lane = threadIdx.x & 31, wid = threadIdx.x >> 5;
  int nw = blockDim.x >> 5;
  if (lane == 0) { sbuf[wid] = a; sbuf[8 + wid] = b; }
  __syncthreads();
  if (wid == 0) {
    a = (lane < nw) ? sbuf[lane] : 0.f;
    b = (lane < nw) ? sbuf[8 + lane] : 0.f;
    for (int o = 4; o > 0; o >>= 1) { a += __shfl_xor(a, o); b += __shfl_xor(b, o); }
    if (lane == 0) { sbuf[16] = a; sbuf[17] = b; }
  }
  __syncthreads();
  a = sbuf[16]; b = sbuf[17];
}

__device__ __forceinline__ h8 pack8(float4 a, float4 b) {
  h8 r;
  r[0] = (_Float16)a.x; r[1] = (_Float16)a.y; r[2] = (_Float16)a.z; r[3] = (_Float16)a.w;
  r[4] = (_Float16)b.x; r[5] = (_Float16)b.y; r[6] = (_Float16)b.z; r[7] = (_Float16)b.w;
  return r;
}

// ---------------------------------------------------------------------------
// Generic WMMA GEMM: C[M x N] = A[M x Kd] * W[N x Kd]^T (+ bias, epilogue)
// block = 256 threads = 8 waves; block tile 64(M) x 128(N);
// wave tile 32x32 -> 4 accumulators, 4 WMMAs per 32-deep K chunk.
// epi: 0 = bias, 1 = bias+GELU(exact), 2 = bias + res[m*ldC+n],
//      3 = bias + res[(m%resS)*ldC+n]   (positional add)
// DUAL: second weight/bias; output = silu(g) * u  (gate/up fusion)
// ---------------------------------------------------------------------------
template <bool DUAL>
__global__ void gemm_kernel(const float* __restrict__ A, const float* __restrict__ W,
                            const float* __restrict__ bias, const float* __restrict__ W2,
                            const float* __restrict__ bias2, const float* __restrict__ res,
                            float* __restrict__ C, int M, int N, int Kd, int ldC,
                            int epi, int resS) {
  __shared__ __attribute__((aligned(16))) _Float16 As[64][40];
  __shared__ __attribute__((aligned(16))) _Float16 Bs[128][40];
  __shared__ __attribute__((aligned(16))) _Float16 Bs2[DUAL ? 128 : 1][40];

  const int t = threadIdx.x;
  const int lane = t & 31, wid = t >> 5;
  const int wmb = (wid >> 2) * 32;   // wave m-base within block tile: 0 / 32
  const int wnb = (wid & 3) * 32;    // wave n-base within block tile: 0..96
  const int m0b = blockIdx.y * 64, n0b = blockIdx.x * 128;

  v8f z = {};
  v8f acc[2][2] = {{z, z}, {z, z}};
  v8f acc2[2][2] = {{z, z}, {z, z}};

  // staging maps: A: 64 rows x 32 k, 8 elems/thread; B: 128 rows x 32 k, 16/thread
  const int arF = t >> 2, akF = (t & 3) * 8;
  const int brF = t >> 1, bkF = (t & 1) * 16;

  const int ml = lane & 15;
  const int khalf = (lane >> 4) * 8;
  const int kb16 = (lane >> 4) * 16;

  for (int kc = 0; kc < Kd; kc += 32) {
    const bool full = (m0b + 64 <= M) && (n0b + 128 <= N) && (kc + 32 <= Kd);
    if (full) {
      // vectorized f32 loads -> f16 LDS stores
      const float4* pa = (const float4*)&A[(size_t)(m0b + arF) * Kd + kc + akF];
      *(h8*)&As[arF][akF] = pack8(pa[0], pa[1]);
      const float4* pb = (const float4*)&W[(size_t)(n0b + brF) * Kd + kc + bkF];
      *(h8*)&Bs[brF][bkF] = pack8(pb[0], pb[1]);
      *(h8*)&Bs[brF][bkF + 8] = pack8(pb[2], pb[3]);
      if (DUAL) {
        const float4* pc = (const float4*)&W2[(size_t)(n0b + brF) * Kd + kc + bkF];
        *(h8*)&Bs2[brF][bkF] = pack8(pc[0], pc[1]);
        *(h8*)&Bs2[brF][bkF + 8] = pack8(pc[2], pc[3]);
      }
      if (kc + 64 <= Kd) {  // gfx1250 prefetch of next K chunk
        __builtin_prefetch(&A[(size_t)(m0b + arF) * Kd + kc + 32 + akF], 0, 1);
        __builtin_prefetch(&W[(size_t)(n0b + brF) * Kd + kc + 32 + bkF], 0, 1);
      }
    } else {
      int gm = m0b + arF;
      for (int j = 0; j < 8; ++j) {
        int k = kc + akF + j;
        As[arF][akF + j] = (_Float16)((gm < M && k < Kd) ? A[(size_t)gm * Kd + k] : 0.f);
      }
      int gn = n0b + brF;
      for (int j = 0; j < 16; ++j) {
        int k = kc + bkF + j;
        bool ok = (gn < N && k < Kd);
        Bs[brF][bkF + j] = (_Float16)(ok ? W[(size_t)gn * Kd + k] : 0.f);
        if (DUAL) Bs2[brF][bkF + j] = (_Float16)(ok ? W2[(size_t)gn * Kd + k] : 0.f);
      }
    }
    __syncthreads();
    {
      union { v16h v; h8 h[2]; } a0, a1, b0, b1;
      a0.h[0] = *(const h8*)&As[wmb + ml][kb16];
      a0.h[1] = *(const h8*)&As[wmb + ml][kb16 + 8];
      a1.h[0] = *(const h8*)&As[wmb + 16 + ml][kb16];
      a1.h[1] = *(const h8*)&As[wmb + 16 + ml][kb16 + 8];
      b0.h[0] = *(const h8*)&Bs[wnb + ml][kb16];
      b0.h[1] = *(const h8*)&Bs[wnb + ml][kb16 + 8];
      b1.h[0] = *(const h8*)&Bs[wnb + 16 + ml][kb16];
      b1.h[1] = *(const h8*)&Bs[wnb + 16 + ml][kb16 + 8];
      acc[0][0] = __builtin_amdgcn_wmma_f32_16x16x32_f16(false, a0.v, false, b0.v, (short)0, acc[0][0], false, false);
      acc[0][1] = __builtin_amdgcn_wmma_f32_16x16x32_f16(false, a0.v, false, b1.v, (short)0, acc[0][1], false, false);
      acc[1][0] = __builtin_amdgcn_wmma_f32_16x16x32_f16(false, a1.v, false, b0.v, (short)0, acc[1][0], false, false);
      acc[1][1] = __builtin_amdgcn_wmma_f32_16x16x32_f16(false, a1.v, false, b1.v, (short)0, acc[1][1], false, false);
      if (DUAL) {
        union { v16h v; h8 h[2]; } c0, c1;
        c0.h[0] = *(const h8*)&Bs2[wnb + ml][kb16];
        c0.h[1] = *(const h8*)&Bs2[wnb + ml][kb16 + 8];
        c1.h[0] = *(const h8*)&Bs2[wnb + 16 + ml][kb16];
        c1.h[1] = *(const h8*)&Bs2[wnb + 16 + ml][kb16 + 8];
        acc2[0][0] = __builtin_amdgcn_wmma_f32_16x16x32_f16(false, a0.v, false, c0.v, (short)0, acc2[0][0], false, false);
        acc2[0][1] = __builtin_amdgcn_wmma_f32_16x16x32_f16(false, a0.v, false, c1.v, (short)0, acc2[0][1], false, false);
        acc2[1][0] = __builtin_amdgcn_wmma_f32_16x16x32_f16(false, a1.v, false, c0.v, (short)0, acc2[1][0], false, false);
        acc2[1][1] = __builtin_amdgcn_wmma_f32_16x16x32_f16(false, a1.v, false, c1.v, (short)0, acc2[1][1], false, false);
      }
    }
    __syncthreads();
  }

  // Epilogue: C-frag lane layout: row = r + 8*(lane>>4), col = lane&15
  for (int i = 0; i < 2; ++i) {
    const int mb = m0b + wmb + i * 16 + 8 * (lane >> 4);
    for (int j = 0; j < 2; ++j) {
      const int n = n0b + wnb + j * 16 + ml;
      if (n >= N) continue;
      float bv = bias ? bias[n] : 0.f;
      float bv2 = (DUAL && bias2) ? bias2[n] : 0.f;
      for (int r = 0; r < 8; ++r) {
        int m = mb + r;
        if (m < M) {
          float v = acc[i][j][r] + bv;
          if (DUAL) {
            float u = acc2[i][j][r] + bv2;
            v = v / (1.f + expf(-v)) * u;             // silu(g) * u
          } else if (epi == 1) {
            v = 0.5f * v * (1.f + erff(v * 0.70710678118f));  // exact GELU
          } else if (epi == 2) {
            v += res[(size_t)m * ldC + n];
          } else if (epi == 3) {
            v += res[(size_t)(m % resS) * ldC + n];
          }
          C[(size_t)m * ldC + n] = v;
        }
      }
    }
  }
}

// ---------------------------------------------------------------------------
// Embedding + positional add:  out[b,s,:] = emb[id[b,s],:] + pos[s,:]
// ---------------------------------------------------------------------------
__global__ void embed_kernel(const int* __restrict__ ids, const float* __restrict__ emb,
                             const float* __restrict__ pos, float* __restrict__ out,
                             int S, int Dd, int total) {
  int i = blockIdx.x * blockDim.x + threadIdx.x;
  if (i >= total) return;
  int d = i % Dd;
  int row = i / Dd;
  int s = row % S;
  out[i] = emb[(size_t)ids[row] * Dd + d] + pos[(size_t)s * Dd + d];
}

// ---------------------------------------------------------------------------
// TemporalFlowCell recurrence: one wave per batch; lane = eigenmode k.
// sr,si kept in registers; K x K matmul by R done with wave shuffles.
// ---------------------------------------------------------------------------
__global__ void scan_kernel(const float* __restrict__ beta, const float* __restrict__ alpha,
                            const float* __restrict__ omega, const float* __restrict__ R,
                            float* __restrict__ srs, int S, int K, int Ktot, int off) {
  int b = blockIdx.x;
  int k = threadIdx.x;
  bool act = k < K;
  float mag = 0.f, cw = 0.f, sw = 0.f;
  if (act) {
    mag = 1.f / (1.f + expf(-alpha[k]));
    cw = cosf(omega[k]);
    sw = sinf(omega[k]);
  }
  float Rc[16];
  for (int j = 0; j < K; ++j) Rc[j] = act ? R[j * K + k] : 0.f;
  float sr = 0.f, si = 0.f;
  size_t base = (size_t)b * S;
  for (int s = 0; s < S; ++s) {
    float bt = act ? beta[(base + s) * Ktot + off + k] : 0.f;
    float nr = mag * (sr * cw - si * sw) + bt;
    float ni = mag * (sr * sw + si * cw);
    float ar = 0.f, ai = 0.f;
    for (int j = 0; j < K; ++j) {
      float nrj = __shfl(nr, j);
      float nij = __shfl(ni, j);
      ar += nrj * Rc[j];
      ai += nij * Rc[j];
    }
    sr = ar; si = ai;
    if (act) srs[(base + s) * Ktot + off + k] = sr;
  }
}

// ---------------------------------------------------------------------------
// Cell output proj + per-cell LayerNorm + accumulate 0.25*LN(y) into accum.
// block = 256 threads per token row; D = 512 (2 elems per thread).
// ---------------------------------------------------------------------------
__global__ void cell_post_kernel(const float* __restrict__ srs, const float* __restrict__ ow,
                                 const float* __restrict__ ob, const float* __restrict__ lnw,
                                 const float* __restrict__ lnb, float* __restrict__ accum,
                                 int K, int Ktot, int off, int first) {
  __shared__ float sred[18];
  __shared__ float sk[16];
  int row = blockIdx.x;
  int t = threadIdx.x;
  if (t < K) sk[t] = srs[(size_t)row * Ktot + off + t];
  __syncthreads();
  float y0 = ob[t], y1 = ob[t + 256];
  for (int k = 0; k < K; ++k) {
    float sv = sk[k];
    y0 += sv * ow[t * K + k];
    y1 += sv * ow[(t + 256) * K + k];
  }
  float s = y0 + y1, sq = y0 * y0 + y1 * y1;
  blockReduce2(s, sq, sred);
  float mean = s * (1.f / 512.f);
  float var = sq * (1.f / 512.f) - mean * mean;
  float rs = rsqrtf(var + 1e-5f);
  float v0 = ((y0 - mean) * rs * lnw[t] + lnb[t]) * 0.25f;
  float v1 = ((y1 - mean) * rs * lnw[t + 256] + lnb[t + 256]) * 0.25f;
  size_t o0 = (size_t)row * 512 + t;
  if (first) { accum[o0] = v0; accum[o0 + 256] = v1; }
  else       { accum[o0] += v0; accum[o0 + 256] += v1; }
}

// ---------------------------------------------------------------------------
// out = LayerNorm(a + b) with weight/bias; b may be null. D = 512, block 256.
// ---------------------------------------------------------------------------
__global__ void add_ln_kernel(const float* __restrict__ a, const float* __restrict__ b,
                              const float* __restrict__ w, const float* __restrict__ bias,
                              float* __restrict__ out) {
  __shared__ float sred[18];
  int row = blockIdx.x;
  int t = threadIdx.x;
  size_t o0 = (size_t)row * 512 + t;
  float x0 = a[o0] + (b ? b[o0] : 0.f);
  float x1 = a[o0 + 256] + (b ? b[o0 + 256] : 0.f);
  float s = x0 + x1, sq = x0 * x0 + x1 * x1;
  blockReduce2(s, sq, sred);
  float mean = s * (1.f / 512.f);
  float var = sq * (1.f / 512.f) - mean * mean;
  float rs = rsqrtf(var + 1e-5f);
  out[o0] = (x0 - mean) * rs * w[t] + bias[t];
  out[o0 + 256] = (x1 - mean) * rs * w[t + 256] + bias[t + 256];
}

// ---------------------------------------------------------------------------
// Flash-style MHA (hd=64). One wave per (b, h, 16-query tile).
// QK^T via WMMA -> 16x512 f32 score panel in LDS -> softmax -> PV via WMMA.
// ---------------------------------------------------------------------------
__global__ void attn_kernel(const float* __restrict__ Q, const float* __restrict__ Km,
                            const float* __restrict__ Vm, float* __restrict__ O,
                            int S, int D, int H) {
  __shared__ __attribute__((aligned(16))) float    Ss[16 * 512];
  __shared__ __attribute__((aligned(16))) _Float16 Ps[16 * 520];

  const int lane = threadIdx.x & 31;
  const int nqt = S >> 4;
  const int qt = blockIdx.x % nqt;
  const int h = (blockIdx.x / nqt) % H;
  const int b = blockIdx.x / (nqt * H);
  const int hd = 64;
  const float* Qb = Q + (size_t)(b * S) * D + h * hd;
  const float* Kb = Km + (size_t)(b * S) * D + h * hd;
  const float* Vb = Vm + (size_t)(b * S) * D + h * hd;
  const int q0 = qt * 16;
  const int ml = lane & 15;
  const int khalf = (lane >> 4) * 8;
  const int kb16 = (lane >> 4) * 16;

  // Hoist Q fragments (hd=64 -> 2 K-chunks of 32)
  v16h aq[2];
  for (int ch = 0; ch < 2; ++ch) {
    union { v16h v; _Float16 e[16]; } u;
    const float* qrow = Qb + (size_t)(q0 + ml) * D + ch * 32;
    for (int i = 0; i < 8; ++i) {
      u.e[i] = (_Float16)qrow[kb16 + i];
      u.e[i + 8] = (_Float16)qrow[kb16 + 8 + i];
    }
    aq[ch] = u.v;
  }

  const float sc = 0.125f;  // 1/sqrt(64)
  const int mrow = 8 * (lane >> 4);
  for (int n0 = 0; n0 < S; n0 += 16) {
    v8f c = {};
    for (int ch = 0; ch < 2; ++ch) {
      union { v16h v; _Float16 e[16]; } ub;
      const float* krow = Kb + (size_t)(n0 + ml) * D + ch * 32 + kb16;
      for (int i = 0; i < 16; ++i) ub.e[i] = (_Float16)krow[i];
      c = __builtin_amdgcn_wmma_f32_16x16x32_f16(false, aq[ch], false, ub.v,
                                                 (short)0, c, false, false);
    }
    for (int r = 0; r < 8; ++r) Ss[(mrow + r) * S + n0 + ml] = c[r] * sc;
  }
  __syncthreads();

  // Softmax: row = lane&15; lanes l and l^16 split the key range.
  {
    int row = ml;
    int half = S >> 1;
    int j0 = (lane >> 4) * half;
    float mx = -1e30f;
    for (int j = 0; j < half; ++j) mx = fmaxf(mx, Ss[row * S + j0 + j]);
    mx = fmaxf(mx, __shfl_xor(mx, 16));
    float sum = 0.f;
    for (int j = 0; j < half; ++j) sum += expf(Ss[row * S + j0 + j] - mx);
    sum += __shfl_xor(sum, 16);
    float inv = 1.f / sum;
    for (int j = 0; j < half; ++j)
      Ps[row * 520 + j0 + j] = (_Float16)(expf(Ss[row * S + j0 + j] - mx) * inv);
  }
  __syncthreads();

  // O = P @ V  (16 x S) x (S x 64): 4 n-tiles, K-loop over keys in 32s
  v8f z = {};
  v8f acc4[4] = {z, z, z, z};
  for (int kc = 0; kc < S; kc += 32) {
    union { v16h v; h8 h[2]; } ua;
    ua.h[0] = *(const h8*)&Ps[ml * 520 + kc + kb16];
    ua.h[1] = *(const h8*)&Ps[ml * 520 + kc + kb16 + 8];
    for (int tt = 0; tt < 4; ++tt) {
      union { v16h v; _Float16 e[16]; } ub;
      for (int i = 0; i < 16; ++i)
        ub.e[i] = (_Float16)Vb[(size_t)(kc + kb16 + i) * D + tt * 16 + ml];
      acc4[tt] = __builtin_amdgcn_wmma_f32_16x16x32_f16(false, ua.v, false, ub.v,
                                                        (short)0, acc4[tt], false, false);
    }
  }
  for (int tt = 0; tt < 4; ++tt)
    for (int r = 0; r < 8; ++r)
      O[(size_t)(b * S + q0 + mrow + r) * D + h * hd + tt * 16 + ml] = acc4[tt][r];
}

// ---------------------------------------------------------------------------
// Mean-pool over sequence: out[b,d] = mean_s x[b,s,d]
// ---------------------------------------------------------------------------
__global__ void pool_kernel(const float* __restrict__ x, float* __restrict__ out,
                            int S, int Dd, int total) {
  int i = blockIdx.x * blockDim.x + threadIdx.x;
  if (i >= total) return;
  int b = i / Dd, d = i % Dd;
  float s = 0.f;
  for (int t = 0; t < S; ++t) s += x[((size_t)b * S + t) * Dd + d];
  out[i] = s / (float)S;
}

// ---------------------------------------------------------------------------
// Host-side orchestration
// ---------------------------------------------------------------------------
struct CellP { const float *alpha, *omega, *W_in, *R, *ow, *ob, *lnw, *lnb; };
struct BlockP {
  CellP c[4];
  const float *gw, *gb, *uw, *ub, *dw, *db, *ln1w, *ln1b, *ln2w, *ln2b;
};

extern "C" void kernel_launch(void* const* d_in, const int* in_sizes, int n_in,
                              void* d_out, int out_size, void* d_ws, size_t ws_size,
                              hipStream_t stream) {
  (void)in_sizes; (void)n_in; (void)out_size; (void)ws_size;

  const int B = 32, S = 512, Dm = 512, VDIM = 768, H = 8;
  const int M = B * S;  // 16384 token rows

  int idx = 0;
  auto nextF = [&]() { return (const float*)d_in[idx++]; };

  const int* text_ids = (const int*)d_in[idx++];
  const float* vision = nextF();
  const float* text_emb = nextF();
  const float* text_pos = nextF();

  auto parseBlock = [&]() {
    BlockP Bp;
    for (int i = 0; i < 4; ++i) {
      Bp.c[i].alpha = nextF(); Bp.c[i].omega = nextF();
      Bp.c[i].W_in = nextF();  Bp.c[i].R = nextF();
      Bp.c[i].ow = nextF();    Bp.c[i].ob = nextF();
      Bp.c[i].lnw = nextF();   Bp.c[i].lnb = nextF();
    }
    Bp.gw = nextF(); Bp.gb = nextF();
    Bp.uw = nextF(); Bp.ub = nextF();
    Bp.dw = nextF(); Bp.db = nextF();
    Bp.ln1w = nextF(); Bp.ln1b = nextF();
    Bp.ln2w = nextF(); Bp.ln2b = nextF();
    return Bp;
  };

  BlockP tb[2] = {parseBlock(), parseBlock()};
  const float* vew = nextF();
  const float* veb = nextF();
  const float* vis_pos = nextF();
  BlockP vb[2] = {parseBlock(), parseBlock()};
  const float* xin_w = nextF();
  const float* xin_b = nextF();
  const float* xow = nextF();
  const float* xob = nextF();
  BlockP fb[2] = {parseBlock(), parseBlock()};
  const float* normw = nextF();
  const float* normb = nextF();
  const float* h1w = nextF();
  const float* h1b = nextF();
  const float* h2w = nextF();
  const float* h2b = nextF();

  // Workspace bump allocator
  float* wsf = (float*)d_ws;
  size_t off = 0;
  auto alloc = [&](size_t n) { float* p = wsf + off; off += n; return p; };
  const size_t MS = (size_t)M * Dm;
  float* ht = alloc(MS);
  float* hv = alloc(MS);
  float* tA = alloc(MS);                    // scratch A (mixed / Q / ffn-out / normed)
  float* tB = alloc(MS);                    // scratch B (ln1 out / K)
  float* G = alloc((size_t)M * 1024);       // fused silu(gate)*up
  float* Vb_ = alloc(MS);
  float* Ob_ = alloc(MS);
  float* beta = alloc((size_t)M * 64);
  float* srs = alloc((size_t)M * 64);
  float* pooled = alloc((size_t)B * Dm);
  float* h1 = alloc((size_t)B * 256);

  auto gemm = [&](const float* A_, const float* W_, const float* b_, float* C_,
                  int M_, int N_, int Kd_, int ldC_, int epi_,
                  const float* res_, int resS_) {
    dim3 g((N_ + 127) / 128, (M_ + 63) / 64);
    gemm_kernel<false><<<g, 256, 0, stream>>>(A_, W_, b_, nullptr, nullptr, res_,
                                              C_, M_, N_, Kd_, ldC_, epi_, resS_);
  };
  auto gemmDual = [&](const float* A_, const float* W_, const float* b_,
                      const float* W2_, const float* b2_, float* C_,
                      int M_, int N_, int Kd_) {
    dim3 g((N_ + 127) / 128, (M_ + 63) / 64);
    gemm_kernel<true><<<g, 256, 0, stream>>>(A_, W_, b_, W2_, b2_, nullptr,
                                             C_, M_, N_, Kd_, N_, 0, 0);
  };

  auto runBlock = [&](float* X, const BlockP& Bp, int K) {
    int Ktot = 4 * K;
    // 1) input projections for 4 cells -> beta[M x Ktot]
    for (int c = 0; c < 4; ++c)
      gemm(X, Bp.c[c].W_in, nullptr, beta + c * K, M, K, Dm, Ktot, 0, nullptr, 0);
    // 2) sequential complex recurrence per (batch, cell)
    for (int c = 0; c < 4; ++c)
      scan_kernel<<<B, 32, 0, stream>>>(beta, Bp.c[c].alpha, Bp.c[c].omega,
                                        Bp.c[c].R, srs, S, K, Ktot, c * K);
    // 3) out-proj + per-cell LN, averaged into tA
    for (int c = 0; c < 4; ++c)
      cell_post_kernel<<<M, 256, 0, stream>>>(srs, Bp.c[c].ow, Bp.c[c].ob,
                                              Bp.c[c].lnw, Bp.c[c].lnb, tA,
                                              K, Ktot, c * K, c == 0 ? 1 : 0);
    // 4) ln1: tB = LN(X + mixed)
    add_ln_kernel<<<M, 256, 0, stream>>>(X, tA, Bp.ln1w, Bp.ln1b, tB);
    // 5) FFN: G = silu(gate(tB)) * up(tB) ; tA = down(G)
    gemmDual(tB, Bp.gw, Bp.gb, Bp.uw, Bp.ub, G, M, 1024, Dm);
    gemm(G, Bp.dw, Bp.db, tA, M, Dm, 1024, Dm, 0, nullptr, 0);
    // 6) ln2: X = LN(tB + ffn)
    add_ln_kernel<<<M, 256, 0, stream>>>(tB, tA, Bp.ln2w, Bp.ln2b, X);
  };

  // ---- text path ----
  embed_kernel<<<(int)((MS + 255) / 256), 256, 0, stream>>>(text_ids, text_emb,
                                                            text_pos, ht, S, Dm,
                                                            (int)MS);
  runBlock(ht, tb[0], 8);
  runBlock(ht, tb[1], 8);

  // ---- vision path: hv = vision @ vew^T + veb + vis_pos ----
  gemm(vision, vew, veb, hv, M, Dm, VDIM, Dm, 3, vis_pos, S);
  runBlock(hv, vb[0], 8);
  runBlock(hv, vb[1], 8);

  // ---- cross attention: Q from ht, K/V from hv ----
  gemm(ht, xin_w, xin_b, tA, M, Dm, Dm, Dm, 0, nullptr, 0);                     // Q
  gemm(hv, xin_w + (size_t)Dm * Dm, xin_b + Dm, tB, M, Dm, Dm, Dm, 0, nullptr, 0);      // K
  gemm(hv, xin_w + (size_t)2 * Dm * Dm, xin_b + 2 * Dm, Vb_, M, Dm, Dm, Dm, 0, nullptr, 0); // V
  attn_kernel<<<B * H * (S / 16), 32, 0, stream>>>(tA, tB, Vb_, Ob_, S, Dm, H);
  // fused = ht + out_proj(attnO)  -> hv (reused)
  gemm(Ob_, xow, xob, hv, M, Dm, Dm, Dm, 2, ht, 0);

  // ---- fusion blocks (K=16) ----
  runBlock(hv, fb[0], 16);
  runBlock(hv, fb[1], 16);

  // ---- final norm, pool, head ----
  add_ln_kernel<<<M, 256, 0, stream>>>(hv, nullptr, normw, normb, tA);
  pool_kernel<<<(B * Dm + 255) / 256, 256, 0, stream>>>(tA, pooled, S, Dm, B * Dm);
  gemm(pooled, h1w, h1b, h1, B, 256, Dm, 256, 1, nullptr, 0);        // GELU head
  gemm(h1, h2w, h2b, (float*)d_out, B, 1000, 256, 1000, 0, nullptr, 0);
}
